// gatedattention_18408229831053
// MI455X (gfx1250) — matrix operand order repaired and checked by reading, not compiled, not run on hardware
//
#include <hip/hip_runtime.h>

#define N_INST 100000
#define DDIM   512
#define CDIM   128
#define KT     (DDIM / 32)   // 16 k-tiles
#define CT     (CDIM / 16)   // 8 column tiles
#define NTILES (N_INST / 16) // 6250 row tiles (exact)

typedef __attribute__((ext_vector_type(16))) __bf16 v16bf;
typedef __attribute__((ext_vector_type(8)))  float  v8f;

// ---- workspace layout (bytes) ----
// [0,       131072)  Wv fragments, bf16, WMMA B-layout
// [131072,  262144)  Wu fragments, bf16, WMMA B-layout
// [262144,  662144)  scores, float[N]
// [662144,  662148)  max key (monotonic uint encoding of float max)
// [662148,  662152)  sum of exp (float)

__device__ __forceinline__ unsigned fkey(float f) {
    unsigned u = __float_as_uint(f);
    return (u & 0x80000000u) ? ~u : (u | 0x80000000u);
}
__device__ __forceinline__ float fdekey(unsigned k) {
    unsigned u = (k & 0x80000000u) ? (k & 0x7FFFFFFFu) : ~k;
    return __uint_as_float(u);
}

// ---------------------------------------------------------------------------
// Kernel 1: repack Wv/Wu (f32 row-major [D,C]) into bf16 WMMA B-fragments.
// B (32x16 bf16) lane layout: lane l -> col = l&15, hi = l>>4;
// vector element e (0..15) -> K = kt*32 + hi*8 + (e&7) + (e>>3)*16.
// Fragment element index = (((kt*8 + ct)*32 + lane)*16 + e)  [bit-packed in idx]
// ---------------------------------------------------------------------------
__global__ void repack_weights(const float* __restrict__ Wv,
                               const float* __restrict__ Wu,
                               __bf16* __restrict__ frag,     // 2*65536 elements
                               unsigned* __restrict__ maxkey,
                               float* __restrict__ sumexp) {
    int idx   = blockIdx.x * blockDim.x + threadIdx.x;  // 0..131071
    int e     = idx & 15;
    int lane  = (idx >> 4) & 31;
    int ct    = (idx >> 9) & 7;
    int kt    = (idx >> 12) & 15;
    int which = idx >> 16;
    int K   = kt * 32 + (lane >> 4) * 8 + (e & 7) + ((e >> 3) << 4);
    int col = ct * 16 + (lane & 15);
    const float* W = which ? Wu : Wv;
    frag[idx] = (__bf16)W[K * CDIM + col];
    if (idx == 0) { *maxkey = 0u; *sumexp = 0.0f; }
}

// ---------------------------------------------------------------------------
// Kernel 2: per-instance gated scores via bf16 WMMA.
// One wave handles 16 rows of x; full C=128 for both Wv and Wu.
// ---------------------------------------------------------------------------
__global__ __launch_bounds__(128) void scores_wmma(
    const float* __restrict__ x,
    const __bf16* __restrict__ vfrag,
    const __bf16* __restrict__ ufrag,
    const float* __restrict__ Wa,
    float* __restrict__ scores) {
    int lane = threadIdx.x & 31;
    int wave = threadIdx.x >> 5;
    int t = blockIdx.x * 4 + wave;
    if (t >= NTILES) return;                    // wave-uniform: EXEC stays all-1s
    int hi = lane >> 4;
    int rowBase = t * 16;
    const float* xrow = x + (size_t)(rowBase + (lane & 15)) * DDIM;

    v8f accV[CT], accU[CT];
    v8f zero = {0.f, 0.f, 0.f, 0.f, 0.f, 0.f, 0.f, 0.f};
#pragma unroll
    for (int c = 0; c < CT; ++c) { accV[c] = zero; accU[c] = zero; }

#pragma unroll 1
    for (int kt = 0; kt < KT; ++kt) {
        int kbase = kt * 32;
        // A 16x32 bf16 layout: lanes hold K = kbase+hi*8+0..7 and kbase+16+hi*8+0..7
        const float4* p1 = (const float4*)(xrow + kbase + hi * 8);
        const float4* p2 = (const float4*)(xrow + kbase + 16 + hi * 8);
        float4 f0 = p1[0], f1 = p1[1], f2 = p2[0], f3 = p2[1];
        v16bf a;
        a[0]  = (__bf16)f0.x; a[1]  = (__bf16)f0.y; a[2]  = (__bf16)f0.z; a[3]  = (__bf16)f0.w;
        a[4]  = (__bf16)f1.x; a[5]  = (__bf16)f1.y; a[6]  = (__bf16)f1.z; a[7]  = (__bf16)f1.w;
        a[8]  = (__bf16)f2.x; a[9]  = (__bf16)f2.y; a[10] = (__bf16)f2.z; a[11] = (__bf16)f2.w;
        a[12] = (__bf16)f3.x; a[13] = (__bf16)f3.y; a[14] = (__bf16)f3.z; a[15] = (__bf16)f3.w;
#pragma unroll
        for (int c = 0; c < CT; ++c) {
            size_t fb = (((size_t)(kt * CT + c)) * 32 + lane) * 16;
            v16bf bv = *(const v16bf*)(vfrag + fb);
            v16bf bu = *(const v16bf*)(ufrag + fb);
            accV[c] = __builtin_amdgcn_wmma_f32_16x16x32_bf16(
                false, a, false, bv, (short)0, accV[c], false, false);
            accU[c] = __builtin_amdgcn_wmma_f32_16x16x32_bf16(
                false, a, false, bu, (short)0, accU[c], false, false);
        }
    }

    // gated score contributions: element (lane, j, ct) -> row j+8*hi, col ct*16+(lane&15)
    float partial[8] = {0.f, 0.f, 0.f, 0.f, 0.f, 0.f, 0.f, 0.f};
#pragma unroll
    for (int c = 0; c < CT; ++c) {
        float wa = Wa[c * 16 + (lane & 15)];
#pragma unroll
        for (int j = 0; j < 8; ++j) {
            float v = tanhf(accV[c][j]);
            float u = 1.0f / (1.0f + expf(-accU[c][j]));
            partial[j] += v * u * wa;
        }
    }
    // reduce over the 16 columns (lanes within each half of the wave32)
#pragma unroll
    for (int j = 0; j < 8; ++j) {
        partial[j] += __shfl_xor(partial[j], 1, 32);
        partial[j] += __shfl_xor(partial[j], 2, 32);
        partial[j] += __shfl_xor(partial[j], 4, 32);
        partial[j] += __shfl_xor(partial[j], 8, 32);
    }
    if ((lane & 15) == 0) {
#pragma unroll
        for (int j = 0; j < 8; ++j)
            scores[rowBase + hi * 8 + j] = partial[j];
    }
}

// ---------------------------------------------------------------------------
// Kernel 3: global max of scores (monotonic-uint atomicMax); zero hs output.
// ---------------------------------------------------------------------------
__global__ void reduce_max(const float* __restrict__ scores,
                           unsigned* __restrict__ maxkey,
                           float* __restrict__ hs) {
    __shared__ float sm[256];
    int tid = threadIdx.x;
    int gid = blockIdx.x * blockDim.x + tid;
    float m = -3.402823466e38f;
    for (int i = gid; i < N_INST; i += gridDim.x * blockDim.x)
        m = fmaxf(m, scores[i]);
    sm[tid] = m;
    __syncthreads();
    for (int s = 128; s > 0; s >>= 1) {
        if (tid < s) sm[tid] = fmaxf(sm[tid], sm[tid + s]);
        __syncthreads();
    }
    if (tid == 0) atomicMax(maxkey, fkey(sm[0]));
    if (gid < DDIM) hs[gid] = 0.0f;
}

// ---------------------------------------------------------------------------
// Kernel 4: unnormalized softmax numerators into d_out[0..N); sum via atomics.
// ---------------------------------------------------------------------------
__global__ void exp_sum(const float* __restrict__ scores,
                        const unsigned* __restrict__ maxkey,
                        float* __restrict__ att_unnorm,
                        float* __restrict__ sumexp) {
    __shared__ float sm[256];
    float mx = fdekey(*maxkey);
    int tid = threadIdx.x;
    int gid = blockIdx.x * blockDim.x + tid;
    float s = 0.0f;
    for (int i = gid; i < N_INST; i += gridDim.x * blockDim.x) {
        float e = expf(scores[i] - mx);
        att_unnorm[i] = e;
        s += e;
    }
    sm[tid] = s;
    __syncthreads();
    for (int st = 128; st > 0; st >>= 1) {
        if (tid < st) sm[tid] += sm[tid + st];
        __syncthreads();
    }
    if (tid == 0) atomicAdd(sumexp, sm[0]);
}

// ---------------------------------------------------------------------------
// Kernel 5: normalize att in place; hs[d] = sum_n att[n]*x[n,d] (chunked).
// ---------------------------------------------------------------------------
__global__ __launch_bounds__(256) void finalize(
    const float* __restrict__ x,
    float* __restrict__ att,
    float* __restrict__ hs,
    const float* __restrict__ sumexp) {
    __shared__ float wsh[256];
    int tid = threadIdx.x;
    int rowBase = blockIdx.x * 256;
    int rows = N_INST - rowBase;
    if (rows > 256) rows = 256;
    float inv = 1.0f / *sumexp;
    if (tid < rows) wsh[tid] = att[rowBase + tid] * inv;
    __syncthreads();
    float acc0 = 0.0f, acc1 = 0.0f;
    for (int r = 0; r < rows; ++r) {
        float w = wsh[r];
        const float* xr = x + (size_t)(rowBase + r) * DDIM;
        acc0 += w * xr[tid];
        acc1 += w * xr[tid + 256];
    }
    atomicAdd(&hs[tid], acc0);
    atomicAdd(&hs[tid + 256], acc1);
    if (tid < rows) att[rowBase + tid] = wsh[tid];
}

// ---------------------------------------------------------------------------
extern "C" void kernel_launch(void* const* d_in, const int* in_sizes, int n_in,
                              void* d_out, int out_size, void* d_ws, size_t ws_size,
                              hipStream_t stream) {
    const float* x  = (const float*)d_in[0];
    const float* Wv = (const float*)d_in[1];
    const float* Wu = (const float*)d_in[2];
    const float* Wa = (const float*)d_in[3];
    float* att = (float*)d_out;           // [0..N)
    float* hs  = att + N_INST;            // [N..N+512)

    char* ws = (char*)d_ws;
    __bf16*   vfrag  = (__bf16*)ws;                 // 65536 elems
    __bf16*   ufrag  = vfrag + 65536;               // 65536 elems
    float*    scores = (float*)(ws + 262144);       // N floats
    unsigned* maxkey = (unsigned*)(ws + 662144);
    float*    sumexp = (float*)(ws + 662148);

    repack_weights<<<512, 256, 0, stream>>>(Wv, Wu, vfrag, maxkey, sumexp);
    scores_wmma<<<(NTILES + 3) / 4, 128, 0, stream>>>(x, vfrag, ufrag, Wa, scores);
    reduce_max<<<391, 256, 0, stream>>>(scores, maxkey, hs);
    exp_sum<<<391, 256, 0, stream>>>(scores, maxkey, att, sumexp);
    finalize<<<(N_INST + 255) / 256, 256, 0, stream>>>(x, att, hs, sumexp);
}
